// GCN_GAT_58110907515030
// MI455X (gfx1250) — matrix operand order repaired and checked by reading, not compiled, not run on hardware
//
#include <hip/hip_runtime.h>
#include <math.h>

// ---------------------------------------------------------------------------
// CDNA5 / gfx1250 GATv2 forward (2 layers)
//   - WMMA bf16 GEMMs (v_wmma_f32_16x16x32_bf16), f32 accumulate
//   - TDM tensor_load_to_lds stages transposed weights into LDS (ds_load reads)
//   - scatter softmax over edges with float atomics
// ---------------------------------------------------------------------------

#ifndef USE_TDM
#define USE_TDM 1
#endif

typedef __bf16 bf16_t;
typedef bf16_t v16bf __attribute__((ext_vector_type(16)));
typedef float  v8f   __attribute__((ext_vector_type(8)));
typedef unsigned int v4u __attribute__((ext_vector_type(4)));
typedef int v8i __attribute__((ext_vector_type(8)));
typedef int v4i __attribute__((ext_vector_type(4)));

struct __align__(16) U4 { unsigned int x, y, z, w; };
union BFrag { U4 q[2]; v16bf v; };

__device__ __forceinline__ unsigned short f2bf(float f) {
    // round-to-nearest-even f32 -> bf16
    unsigned int u = __float_as_uint(f);
    u += 0x7FFFu + ((u >> 16) & 1u);
    return (unsigned short)(u >> 16);
}

__device__ __forceinline__ void atomicMaxF(float* addr, float v) {
    // ordered-int float max: works for mixed signs with -inf init
    if (v >= 0.0f) atomicMax((int*)addr, __float_as_int(v));
    else           atomicMin((unsigned int*)addr, __float_as_uint(v));
}

// --------------------------- small utility kernels --------------------------

__global__ void k_f32_to_bf16(const float* __restrict__ in,
                              unsigned short* __restrict__ out, long long n) {
    long long i = (long long)blockIdx.x * blockDim.x + threadIdx.x;
    if (i < n) out[i] = f2bf(in[i]);
}

// W is [K][Nout] row-major f32; produce Bt [Nout][K] bf16 (transposed weights)
__global__ void k_transpose_bf16(const float* __restrict__ W,
                                 unsigned short* __restrict__ Bt,
                                 int K, int Nout) {
    int i = blockIdx.x * blockDim.x + threadIdx.x;
    if (i < K * Nout) {
        int k = i / Nout, n = i % Nout;
        Bt[(size_t)n * K + k] = f2bf(W[i]);
    }
}

__global__ void k_fill(float* __restrict__ p, long long n, float v) {
    long long i = (long long)blockIdx.x * blockDim.x + threadIdx.x;
    if (i < n) p[i] = v;
}

// ----------------------------- WMMA bf16 GEMM -------------------------------
// C[M][Nout] = A[M][128] (bf16) * Bt[Nout][128]^T (bf16) + bias,  f32 accum.
// Bt (whole weight matrix, <=32KB bf16) is DMA'd into LDS once per block via
// the Tensor Data Mover; waves then pull B fragments with ds_load_b128.
// One wave -> one 16x16 tile; K=128 = 4 chunks of v_wmma_f32_16x16x32_bf16.
__global__ __launch_bounds__(128) void k_gemm_bf16_wmma(
    const unsigned short* __restrict__ A,
    const unsigned short* __restrict__ Bt,
    const float* __restrict__ bias,
    float* __restrict__ C,
    int M, int Nout)
{
    __shared__ unsigned short lds_bt[128 * 128];   // 32 KB

    const int lane    = threadIdx.x & 31;
    const int wave    = threadIdx.x >> 5;
    const int l16     = lane & 15;
    const int half    = lane >> 4;          // lane group 0-15 / 16-31
    const int rowbase = blockIdx.x * 16;

    // The TDM writes LDS behind the compiler's back; this (never-executed)
    // store stops LLVM from proving lds_bt is write-free and folding every
    // ds_load of it to undef.  Nout is a runtime arg, so it can't be proven
    // dead at compile time.
    if (__builtin_expect(Nout == -1, 0)) lds_bt[threadIdx.x] = (unsigned short)M;

#if USE_TDM && __has_builtin(__builtin_amdgcn_tensor_load_to_lds)
    // ---- TDM: one wave issues a 2-D descriptor DMA for Bt[Nout][128] ----
    if (wave == 0) {
        unsigned long long ga = (unsigned long long)(uintptr_t)Bt;
        // LDS byte offset = low 32 bits of the generic shared address
        // (flat aperture rule); the ptrtoint also
        // makes lds_bt escape into the intrinsic.
        unsigned ldsOff = (unsigned)(unsigned long long)(uintptr_t)&lds_bt[0];
        // D# group 0: count=1, lds_addr, global_addr, type=2 ("image")
        v4u g0 = { 1u,
                   ldsOff,
                   (unsigned)(ga & 0xFFFFFFFFu),
                   (unsigned)((ga >> 32) & 0x01FFFFFFu) | (2u << 30) };
        // D# group 1: data_size=2B; tensor_dim0=128, tensor_dim1=Nout;
        // tile_dim0=128, tile_dim1=Nout; tensor_dim0_stride=128
        v8i g1 = { (int)(1u << 16),
                   (int)(128u << 16),
                   (int)((unsigned)(Nout & 0xFFFF) << 16),
                   (int)(128u << 16),
                   (int)(Nout & 0xFFFF),
                   128, 0, 0 };
        v4i gz4 = { 0, 0, 0, 0 };
        v8i gz8 = { 0, 0, 0, 0, 0, 0, 0, 0 };
        __builtin_amdgcn_tensor_load_to_lds(g0, g1, gz4, gz4, gz8, 0);
    }
#else
    // ---- fallback: cooperative copy global -> LDS (16B chunks) ----
    {
        const int chunks = Nout * 16;              // (Nout*128)/8 ushorts per 16B
        const U4* src16 = (const U4*)Bt;
        U4* dst16 = (U4*)lds_bt;
        for (int i = threadIdx.x; i < chunks; i += 128) dst16[i] = src16[i];
    }
#endif

    // ---- A fragments from global (overlaps the B-tile DMA) ----
    int arow = rowbase + l16;
    if (arow >= M) arow = M - 1;
    const unsigned short* aPtr = A + (size_t)arow * 128;
    BFrag a[4];
#pragma unroll
    for (int kc = 0; kc < 4; ++kc) {
        const unsigned short* p = aPtr + kc * 32 + half * 8;
        a[kc].q[0] = *(const U4*)(p);
        a[kc].q[1] = *(const U4*)(p + 16);
    }

#if USE_TDM && __has_builtin(__builtin_amdgcn_tensor_load_to_lds)
    if (wave == 0) __builtin_amdgcn_s_wait_tensorcnt(0);
#endif
    __syncthreads();

    const int NT = Nout >> 4;
    for (int ct = wave; ct < NT; ct += 4) {
        // B layout: lane = column n; per lane K = half*8 + [0,8) and 16 + half*8 + [0,8)
        const unsigned short* bPtr = &lds_bt[(ct * 16 + l16) * 128 + half * 8];
        v8f acc = {};
#pragma unroll
        for (int kc = 0; kc < 4; ++kc) {
            BFrag b;
            b.q[0] = *(const U4*)(bPtr + kc * 32);      // ds_load_b128
            b.q[1] = *(const U4*)(bPtr + kc * 32 + 16); // ds_load_b128
            acc = __builtin_amdgcn_wmma_f32_16x16x32_bf16(
                      false, a[kc].v, false, b.v, (short)0, acc, false, false);
        }
        const int col  = ct * 16 + l16;
        const float bv = bias[col];
        const int row0 = rowbase + half * 8;   // C/D layout: M = half*8 + r
        if (rowbase + 16 <= M) {               // hot path: branchless stores
#pragma unroll
            for (int r = 0; r < 8; ++r)
                C[(size_t)(row0 + r) * Nout + col] = acc[r] + bv;
        } else {
#pragma unroll
            for (int r = 0; r < 8; ++r)
                if (row0 + r < M) C[(size_t)(row0 + r) * Nout + col] = acc[r] + bv;
        }
    }
}

// ----------------------------- edge-wise kernels ----------------------------
// Edge id e < E: real edge; e >= E: self-loop (src=dst=e-E).

template<int HEADS, int CH>
__global__ void k_edge_scores(
    const float* __restrict__ xl, const float* __restrict__ xr,
    const float* __restrict__ att,
    const int* __restrict__ src, const int* __restrict__ dst,
    int E, int N, float* __restrict__ ebuf, float* __restrict__ emax)
{
    long long gid = (long long)blockIdx.x * blockDim.x + threadIdx.x;
    long long total = (long long)(E + N) * HEADS;
    if (gid >= total) return;
    int e = (int)(gid / HEADS);
    int h = (int)(gid % HEADS);
    int s, d;
    if (e < E) { s = src[e]; d = dst[e]; } else { s = d = e - E; }
    const float* pl = xl + ((size_t)s * HEADS + h) * CH;
    const float* pr = xr + ((size_t)d * HEADS + h) * CH;
    __builtin_prefetch(pl, 0, 0);              // global_prefetch_b8
    __builtin_prefetch(pr, 0, 0);
    const float* pa = att + h * CH;
    float acc = 0.f;
#pragma unroll
    for (int c = 0; c < CH; ++c) {
        float m = pl[c] + pr[c];
        m = (m > 0.f) ? m : 0.2f * m;          // leaky_relu(0.2)
        acc += m * pa[c];
    }
    ebuf[(size_t)e * HEADS + h] = acc;
    atomicMaxF(&emax[(size_t)d * HEADS + h], acc);
}

template<int HEADS>
__global__ void k_edge_exp(
    const int* __restrict__ src, const int* __restrict__ dst,
    int E, int N, float* __restrict__ ebuf,
    const float* __restrict__ emax, float* __restrict__ denom)
{
    long long gid = (long long)blockIdx.x * blockDim.x + threadIdx.x;
    long long total = (long long)(E + N) * HEADS;
    if (gid >= total) return;
    int e = (int)(gid / HEADS);
    int h = (int)(gid % HEADS);
    int d = (e < E) ? dst[e] : e - E;
    size_t idx = (size_t)e * HEADS + h;
    float ee = expf(ebuf[idx] - emax[(size_t)d * HEADS + h]);
    ebuf[idx] = ee;                            // overwrite score with exp value
    atomicAdd(&denom[(size_t)d * HEADS + h], ee);
}

template<int HEADS, int CH>
__global__ void k_edge_aggregate(
    const float* __restrict__ xl,
    const int* __restrict__ src, const int* __restrict__ dst,
    int E, int N, const float* __restrict__ ebuf,
    const float* __restrict__ denom, float* __restrict__ agg)
{
    constexpr int HC = HEADS * CH;
    long long gid = (long long)blockIdx.x * blockDim.x + threadIdx.x;
    long long total = (long long)(E + N) * HC;
    if (gid >= total) return;
    int e = (int)(gid / HC);
    int c = (int)(gid % HC);
    int h = c / CH;
    int s, d;
    if (e < E) { s = src[e]; d = dst[e]; } else { s = d = e - E; }
    float alpha = ebuf[(size_t)e * HEADS + h] /
                  (denom[(size_t)d * HEADS + h] + 1e-16f);
    atomicAdd(&agg[(size_t)d * HC + c], xl[(size_t)s * HC + c] * alpha);
}

// layer-1 epilogue: elu(agg + bias) -> bf16 input for layer 2
__global__ void k_elu_bias_bf16(const float* __restrict__ agg,
                                const float* __restrict__ bias,
                                unsigned short* __restrict__ outbf,
                                long long n, int HC)
{
    long long i = (long long)blockIdx.x * blockDim.x + threadIdx.x;
    if (i >= n) return;
    float v = agg[i] + bias[(int)(i % HC)];
    v = (v > 0.f) ? v : (expf(v) - 1.f);       // elu
    outbf[i] = f2bf(v);
}

// final: h = agg2 + bias2 ; out = log_softmax(h) ; pred = argmax
__global__ void k_final(const float* __restrict__ agg2,
                        const float* __restrict__ bias2,
                        float* __restrict__ h_out, float* __restrict__ ls_out,
                        int* __restrict__ pred, int N)
{
    int n = blockIdx.x * blockDim.x + threadIdx.x;
    if (n >= N) return;
    float v[64];
    float mx = -INFINITY; int am = 0;
#pragma unroll
    for (int c = 0; c < 64; ++c) {
        v[c] = agg2[(size_t)n * 64 + c] + bias2[c];
        if (v[c] > mx) { mx = v[c]; am = c; }
    }
    float s = 0.f;
#pragma unroll
    for (int c = 0; c < 64; ++c) s += expf(v[c] - mx);
    float lse = mx + logf(s);
#pragma unroll
    for (int c = 0; c < 64; ++c) {
        h_out[(size_t)n * 64 + c]  = v[c];
        ls_out[(size_t)n * 64 + c] = v[c] - lse;
    }
    pred[n] = am;
}

// ------------------------------- launcher -----------------------------------

extern "C" void kernel_launch(void* const* d_in, const int* in_sizes, int n_in,
                              void* d_out, int out_size, void* d_ws, size_t ws_size,
                              hipStream_t stream)
{
    const float* x     = (const float*)d_in[0];
    const int*   ei    = (const int*)d_in[1];     // int32 (jax default, x64 off)
    const float* Wl1   = (const float*)d_in[2];
    const float* bl1   = (const float*)d_in[3];
    const float* Wr1   = (const float*)d_in[4];
    const float* br1   = (const float*)d_in[5];
    const float* att1  = (const float*)d_in[6];
    const float* bias1 = (const float*)d_in[7];
    const float* Wl2   = (const float*)d_in[8];
    const float* bl2   = (const float*)d_in[9];
    const float* Wr2   = (const float*)d_in[10];
    const float* br2   = (const float*)d_in[11];
    const float* att2  = (const float*)d_in[12];
    const float* bias2 = (const float*)d_in[13];

    const int N  = in_sizes[0] / 128;
    const int E  = in_sizes[1] / 2;
    const int ET = E + N;                         // edges + self loops
    const int* src = ei;
    const int* dst = ei + E;

    // ---- workspace bump allocator (with reuse); peak ~133MB, fits in L2 ----
    char* ws = (char*)d_ws;
    size_t off = 0;
    auto alloc = [&](size_t bytes) -> void* {
        void* p = ws + off;
        off += (bytes + 255) & ~(size_t)255;
        return p;
    };
    unsigned short* xbf  = (unsigned short*)alloc((size_t)N * 128 * 2);
    unsigned short* w1lt = (unsigned short*)alloc(128 * 128 * 2);
    unsigned short* w1rt = (unsigned short*)alloc(128 * 128 * 2);
    unsigned short* w2lt = (unsigned short*)alloc(64 * 128 * 2);
    unsigned short* w2rt = (unsigned short*)alloc(64 * 128 * 2);
    float* xl1   = (float*)alloc((size_t)N * 128 * 4);   // reused as xl2
    float* xr1   = (float*)alloc((size_t)N * 128 * 4);   // reused as xr2
    float* ebuf  = (float*)alloc((size_t)ET * 8 * 4);    // reused for layer 2
    float* emax  = (float*)alloc((size_t)N * 8 * 4);
    float* denom = (float*)alloc((size_t)N * 8 * 4);
    float* agg1  = (float*)alloc((size_t)N * 128 * 4);   // reused as agg2
    unsigned short* h1bf = (unsigned short*)alloc((size_t)N * 128 * 2);

    const int TB = 256;
    auto nb = [](long long n, int tb) { return (unsigned)((n + tb - 1) / tb); };
    const unsigned ggrid = (unsigned)((N + 15) / 16);

    // ---- precision conversion ----
    long long nx = (long long)N * 128;
    k_f32_to_bf16<<<nb(nx, TB), TB, 0, stream>>>(x, xbf, nx);
    k_transpose_bf16<<<nb(128 * 128, TB), TB, 0, stream>>>(Wl1, w1lt, 128, 128);
    k_transpose_bf16<<<nb(128 * 128, TB), TB, 0, stream>>>(Wr1, w1rt, 128, 128);
    k_transpose_bf16<<<nb(128 * 64, TB), TB, 0, stream>>>(Wl2, w2lt, 128, 64);
    k_transpose_bf16<<<nb(128 * 64, TB), TB, 0, stream>>>(Wr2, w2rt, 128, 64);

    // ---- layer 1: GEMMs (WMMA + TDM-staged weights) ----
    k_gemm_bf16_wmma<<<ggrid, 128, 0, stream>>>(xbf, w1lt, bl1, xl1, N, 128);
    k_gemm_bf16_wmma<<<ggrid, 128, 0, stream>>>(xbf, w1rt, br1, xr1, N, 128);

    // ---- layer 1: scatter softmax + aggregation ----
    k_fill<<<nb((long long)N * 8, TB), TB, 0, stream>>>(emax, (long long)N * 8, -INFINITY);
    k_fill<<<nb((long long)N * 8, TB), TB, 0, stream>>>(denom, (long long)N * 8, 0.f);
    k_fill<<<nb((long long)N * 128, TB), TB, 0, stream>>>(agg1, (long long)N * 128, 0.f);

    k_edge_scores<8, 16><<<nb((long long)ET * 8, TB), TB, 0, stream>>>(
        xl1, xr1, att1, src, dst, E, N, ebuf, emax);
    k_edge_exp<8><<<nb((long long)ET * 8, TB), TB, 0, stream>>>(
        src, dst, E, N, ebuf, emax, denom);
    k_edge_aggregate<8, 16><<<nb((long long)ET * 128, TB), TB, 0, stream>>>(
        xl1, src, dst, E, N, ebuf, denom, agg1);

    k_elu_bias_bf16<<<nb((long long)N * 128, TB), TB, 0, stream>>>(
        agg1, bias1, h1bf, (long long)N * 128, 128);

    // ---- layer 2: GEMMs (WMMA), reuse xl1/xr1 as xl2/xr2 ----
    k_gemm_bf16_wmma<<<ggrid, 128, 0, stream>>>(h1bf, w2lt, bl2, xl1, N, 64);
    k_gemm_bf16_wmma<<<ggrid, 128, 0, stream>>>(h1bf, w2rt, br2, xr1, N, 64);

    // ---- layer 2: scatter softmax + aggregation (H=1, C=64) ----
    k_fill<<<nb((long long)N, TB), TB, 0, stream>>>(emax, (long long)N, -INFINITY);
    k_fill<<<nb((long long)N, TB), TB, 0, stream>>>(denom, (long long)N, 0.f);
    k_fill<<<nb((long long)N * 64, TB), TB, 0, stream>>>(agg1, (long long)N * 64, 0.f);

    k_edge_scores<1, 64><<<nb((long long)ET, TB), TB, 0, stream>>>(
        xl1, xr1, att2, src, dst, E, N, ebuf, emax);
    k_edge_exp<1><<<nb((long long)ET, TB), TB, 0, stream>>>(
        src, dst, E, N, ebuf, emax, denom);
    k_edge_aggregate<1, 64><<<nb((long long)ET * 64, TB), TB, 0, stream>>>(
        xl1, src, dst, E, N, ebuf, denom, agg1);

    // ---- final: bias + log_softmax + argmax ----
    float* Ho = (float*)d_out;                    // h            [N,64]
    float* Lo = Ho + (size_t)N * 64;              // log_softmax  [N,64]
    int*   Po = (int*)(Ho + (size_t)2 * N * 64);  // predictions  [N] (int32)
    k_final<<<nb(N, TB), TB, 0, stream>>>(agg1, bias2, Ho, Lo, Po, N);
}
